// GatedLinearUnit_68582037782943
// MI455X (gfx1250) — matrix operand order repaired and checked by reading, not compiled, not run on hardware
//
#include <hip/hip_runtime.h>
#include <cstdint>

// ---------------------------------------------------------------------------
// Quantized GLU block for MI455X (gfx1250), wave32 + v_wmma_f32_16x16x32_f16.
//   out0 = relu(s_x*(q_x@wg^T)+bg) * (s_x*(q_x@w1^T)+b1)        [GEMM1 fused]
//   out1 = out0 @ rot            [GEMM2: async-to-LDS double-buffered, absmax]
//   res  = s_o*(q_out1 @ w2^T) + b2                              [GEMM3]
// Workspace: [0..255] scalars | rotT16 128MB | out0 f16 128MB | out1 f16 128MB
// ---------------------------------------------------------------------------

#define EMB 2048
#define HID 8192
#define TOK 8192
#define QMAX 127.0f
#define LDH 40   // LDS row stride (halves) for 32-wide K tiles
#define LDH2 72  // LDS row stride (halves) for 64-wide K tiles (144B = 9*16B)

typedef __attribute__((ext_vector_type(16))) _Float16 v16h;
typedef __attribute__((ext_vector_type(8)))  _Float16 h8;
typedef __attribute__((ext_vector_type(4)))  _Float16 h4;
typedef __attribute__((ext_vector_type(8)))  float    v8f;

union Frag16 { v16h v; h8 h[2]; };

__device__ __forceinline__ v8f wmma16(const Frag16& a, const Frag16& b, v8f c) {
  return __builtin_amdgcn_wmma_f32_16x16x32_f16(false, a.v, false, b.v,
                                                (short)0, c, false, false);
}

// A fragment from LDS tile (row-major, `stride` halves/row), k-sub-tile at kk.
// lane holds row m=base+(l&15); elems 0..7 = K[kk+kb..), 8..15 = K[kk+16+kb..),
// kb=(l>>4)*8.  (ISA 16-bit A 16x32 layout)
__device__ __forceinline__ Frag16 load_afrag(const _Float16* lds, int stride,
                                             int mbase, int kk, int lane) {
  Frag16 f;
  const int m  = mbase + (lane & 15);
  const int kb = kk + ((lane >> 4) << 3);
  f.h[0] = *(const h8*)(lds + m * stride + kb);
  f.h[1] = *(const h8*)(lds + m * stride + kb + 16);
  return f;
}

// B fragment: lane holds col n=base+(l&15); elems 0..7 = K[kk+kb..), 8..15 = K[kk+16+kb..),
// kb=(l>>4)*8.  LDS holds B as Wt[N][K] row-major.
__device__ __forceinline__ Frag16 load_bfrag(const _Float16* lds, int stride,
                                             int nbase, int kk, int lane) {
  Frag16 f;
  const int n  = nbase + (lane & 15);
  const int kb = kk + ((lane >> 4) << 3);
  f.h[0] = *(const h8*)(lds + n * stride + kb);
  f.h[1] = *(const h8*)(lds + n * stride + kb + 16);
  return f;
}

__device__ __forceinline__ float qclip(float v, float inv_s) {
  return fminf(fmaxf(rintf(v * inv_s), -QMAX), QMAX);
}

// gfx1250 async copy: 16B global -> LDS per lane, tracked by ASYNCcnt.
__device__ __forceinline__ void async_b128(uint32_t lds_addr, const void* gaddr) {
  asm volatile("global_load_async_to_lds_b128 %0, %1, off"
               :: "v"(lds_addr), "v"(gaddr) : "memory");
}
#define S_WAIT_ASYNC(n) asm volatile("s_wait_asynccnt " #n ::: "memory")

__device__ __forceinline__ uint32_t lds_off(const void* p) {
  return (uint32_t)(uintptr_t)p;   // LDS aperture: addr[31:0] = LDS offset
}

// ---------------------------------------------------------------------------
__global__ __launch_bounds__(32) void glu_init_scalars(unsigned* scal) {
  if (threadIdx.x < 2) scal[threadIdx.x] = 0u;
}

__global__ __launch_bounds__(256) void glu_absmax_f32(const float* __restrict__ p,
                                                      size_t n4, unsigned* slot) {
  float lm = 0.0f;
  for (size_t i = (size_t)blockIdx.x * blockDim.x + threadIdx.x; i < n4;
       i += (size_t)gridDim.x * blockDim.x) {
    float4 v = ((const float4*)p)[i];
    lm = fmaxf(lm, fmaxf(fmaxf(fabsf(v.x), fabsf(v.y)),
                         fmaxf(fabsf(v.z), fabsf(v.w))));
  }
  __shared__ float red[256];
  red[threadIdx.x] = lm;
  __syncthreads();
  for (int st = 128; st > 0; st >>= 1) {
    if ((int)threadIdx.x < st)
      red[threadIdx.x] = fmaxf(red[threadIdx.x], red[threadIdx.x + st]);
    __syncthreads();
  }
  if (threadIdx.x == 0) atomicMax(slot, __float_as_uint(red[0]));
}

// rot [K=HID][N=HID] f32 row-major -> rotT16 [N][K] f16 row-major
__global__ __launch_bounds__(256) void glu_rot_transpose(const float* __restrict__ rot,
                                                         _Float16* __restrict__ rotT) {
  __shared__ float t[32][33];
  const int tx = threadIdx.x & 31, ty = threadIdx.x >> 5;   // 32x8
  const int n0 = blockIdx.x * 32, k0 = blockIdx.y * 32;
#pragma unroll
  for (int i = 0; i < 4; ++i)
    t[ty + i * 8][tx] = rot[(size_t)(k0 + ty + i * 8) * HID + n0 + tx];
  __syncthreads();
#pragma unroll
  for (int i = 0; i < 4; ++i)
    rotT[(size_t)(n0 + ty + i * 8) * HID + k0 + tx] = (_Float16)t[tx][ty + i * 8];
}

// ---------------------------------------------------------------------------
// GEMM1 fused gate/up: block tile 128(M) x 64(N), K-step 32, 8 waves @ 32x32.
__global__ __launch_bounds__(256)
void glu_gemm1(const float* __restrict__ x,  const float* __restrict__ wg,
               const float* __restrict__ w1, const float* __restrict__ bg,
               const float* __restrict__ b1, const float* __restrict__ scal,
               _Float16* __restrict__ out0) {
  __shared__ __align__(16) _Float16 As[128 * LDH];
  __shared__ __align__(16) _Float16 Bg[64 * LDH];
  __shared__ __align__(16) _Float16 Bu[64 * LDH];

  const int tid  = threadIdx.x;
  const int lane = tid & 31, wv = tid >> 5;
  const int m0 = blockIdx.x * 128, n0 = blockIdx.y * 64;
  const int wm = (wv >> 1) * 32, wn = (wv & 1) * 32;

  const float s = fmaxf(scal[0] * (1.0f / QMAX), 1e-8f);  // scal holds absmax bits
  const float inv_s = 1.0f / s;

  v8f accg[2][2] = {}, accu[2][2] = {};

  for (int k0 = 0; k0 < EMB; k0 += 32) {
    // A: x f32 128x32 -> quantize -> f16 LDS
#pragma unroll
    for (int i = 0; i < 4; ++i) {
      const int idx = tid + i * 256, r = idx >> 3, c = idx & 7;
      float4 v = *(const float4*)(x + (size_t)(m0 + r) * EMB + k0 + c * 4);
      h4 q;
      q.x = (_Float16)qclip(v.x, inv_s); q.y = (_Float16)qclip(v.y, inv_s);
      q.z = (_Float16)qclip(v.z, inv_s); q.w = (_Float16)qclip(v.w, inv_s);
      *(h4*)(As + r * LDH + c * 4) = q;
    }
    // B: wg/w1 rows [n0..n0+64) x 32 f32 -> f16 LDS
#pragma unroll
    for (int i = 0; i < 2; ++i) {
      const int idx = tid + i * 256, r = idx >> 3, c = idx & 7;
      float4 vg = *(const float4*)(wg + (size_t)(n0 + r) * EMB + k0 + c * 4);
      float4 vu = *(const float4*)(w1 + (size_t)(n0 + r) * EMB + k0 + c * 4);
      h4 hg, hu;
      hg.x = (_Float16)vg.x; hg.y = (_Float16)vg.y; hg.z = (_Float16)vg.z; hg.w = (_Float16)vg.w;
      hu.x = (_Float16)vu.x; hu.y = (_Float16)vu.y; hu.z = (_Float16)vu.z; hu.w = (_Float16)vu.w;
      *(h4*)(Bg + r * LDH + c * 4) = hg;
      *(h4*)(Bu + r * LDH + c * 4) = hu;
    }
    __syncthreads();

    Frag16 a[2], fg[2], fu[2];
#pragma unroll
    for (int tm = 0; tm < 2; ++tm) a[tm] = load_afrag(As, LDH, wm + tm * 16, 0, lane);
#pragma unroll
    for (int tn = 0; tn < 2; ++tn) {
      fg[tn] = load_bfrag(Bg, LDH, wn + tn * 16, 0, lane);
      fu[tn] = load_bfrag(Bu, LDH, wn + tn * 16, 0, lane);
    }
#pragma unroll
    for (int tm = 0; tm < 2; ++tm)
#pragma unroll
      for (int tn = 0; tn < 2; ++tn) {
        accg[tm][tn] = wmma16(a[tm], fg[tn], accg[tm][tn]);
        accu[tm][tn] = wmma16(a[tm], fu[tn], accu[tm][tn]);
      }
    __syncthreads();
  }

#pragma unroll
  for (int tm = 0; tm < 2; ++tm)
#pragma unroll
    for (int tn = 0; tn < 2; ++tn) {
      const int n  = n0 + wn + tn * 16 + (lane & 15);
      const float bgn = bg[n], b1n = b1[n];
      const int mb = m0 + wm + tm * 16 + ((lane >> 4) << 3);
#pragma unroll
      for (int r = 0; r < 8; ++r) {
        float g = fmaxf(s * accg[tm][tn][r] + bgn, 0.0f);
        float u = s * accu[tm][tn][r] + b1n;
        out0[(size_t)(mb + r) * HID + n] = (_Float16)(g * u);
      }
    }
}

// ---------------------------------------------------------------------------
// GEMM2: out1 = out0 @ rot.  128x128 tile, K-step 64, 8 waves @ 32x64.
// A,B tiles copied global->LDS with gfx1250 async-to-LDS ops, double-buffered.
__global__ __launch_bounds__(256)
void glu_gemm2(const _Float16* __restrict__ out0, const _Float16* __restrict__ rotT,
               _Float16* __restrict__ out1, unsigned* __restrict__ amax_slot) {
  __shared__ __align__(16) _Float16 As[2][128 * LDH2];
  __shared__ __align__(16) _Float16 Bs[2][128 * LDH2];
  __shared__ float red[256];

  const int tid = threadIdx.x;
  const int lane = tid & 31, wv = tid >> 5;
  const int m0 = blockIdx.x * 128, n0 = blockIdx.y * 128;
  const int wm = (wv & 3) * 32, wn = (wv >> 2) * 64;

  const _Float16* gA = out0 + (size_t)m0 * HID;   // [128][HID] slab
  const _Float16* gB = rotT + (size_t)n0 * HID;

  v8f acc[2][4] = {};

  // Issue one 128x64-half tile pair (A+B) as 8 async b128 copies per thread.
  auto stage = [&](int buf, int k0) {
#pragma unroll
    for (int i = 0; i < 4; ++i) {
      const int idx = tid + i * 256, r = idx >> 3, c = idx & 7;  // 8 chunks/row
      async_b128(lds_off(&As[buf][r * LDH2 + c * 8]),
                 gA + (size_t)r * HID + k0 + c * 8);
      async_b128(lds_off(&Bs[buf][r * LDH2 + c * 8]),
                 gB + (size_t)r * HID + k0 + c * 8);
    }
  };

  constexpr int KT = HID / 64;
  int buf = 0;
  stage(0, 0);
  for (int kt = 0; kt < KT; ++kt) {
    if (kt + 1 < KT) {
      stage(buf ^ 1, (kt + 1) * 64);      // prefetch next tile (8 ops in flight)
      S_WAIT_ASYNC(8);                    // current tile's 8 ops have completed
    } else {
      S_WAIT_ASYNC(0);
    }
    __syncthreads();                      // all lanes' async data visible

#pragma unroll
    for (int kk = 0; kk < 64; kk += 32) {
      Frag16 a[2], b[4];
#pragma unroll
      for (int tm = 0; tm < 2; ++tm)
        a[tm] = load_afrag(As[buf], LDH2, wm + tm * 16, kk, lane);
#pragma unroll
      for (int tn = 0; tn < 4; ++tn)
        b[tn] = load_bfrag(Bs[buf], LDH2, wn + tn * 16, kk, lane);
#pragma unroll
      for (int tm = 0; tm < 2; ++tm)
#pragma unroll
        for (int tn = 0; tn < 4; ++tn)
          acc[tm][tn] = wmma16(a[tm], b[tn], acc[tm][tn]);
    }
    __syncthreads();                      // safe to overwrite `buf` next iter
    buf ^= 1;
  }

  float lmax = 0.0f;
#pragma unroll
  for (int tm = 0; tm < 2; ++tm)
#pragma unroll
    for (int tn = 0; tn < 4; ++tn) {
      const int n  = n0 + wn + tn * 16 + (lane & 15);
      const int mb = m0 + wm + tm * 16 + ((lane >> 4) << 3);
#pragma unroll
      for (int r = 0; r < 8; ++r) {
        float v = acc[tm][tn][r];
        lmax = fmaxf(lmax, fabsf(v));
        out1[(size_t)(mb + r) * HID + n] = (_Float16)v;
      }
    }
  red[tid] = lmax;
  __syncthreads();
  for (int st = 128; st > 0; st >>= 1) {
    if (tid < st) red[tid] = fmaxf(red[tid], red[tid + st]);
    __syncthreads();
  }
  if (tid == 0) atomicMax(amax_slot, __float_as_uint(red[0]));
}

// ---------------------------------------------------------------------------
// GEMM3: res = fake_quant(out1) @ w2^T + b2.  Quantize folded into A staging.
__global__ __launch_bounds__(256)
void glu_gemm3(const _Float16* __restrict__ out1, const float* __restrict__ w2,
               const float* __restrict__ b2, const float* __restrict__ scal,
               float* __restrict__ out) {
  __shared__ __align__(16) _Float16 As[128 * LDH];
  __shared__ __align__(16) _Float16 Bs[128 * LDH];

  const int tid = threadIdx.x;
  const int lane = tid & 31, wv = tid >> 5;
  const int m0 = blockIdx.x * 128, n0 = blockIdx.y * 128;
  const int wm = (wv & 3) * 32, wn = (wv >> 2) * 64;

  const float s = fmaxf(scal[0] * (1.0f / QMAX), 1e-8f);
  const float inv_s = 1.0f / s;

  v8f acc[2][4] = {};

  for (int k0 = 0; k0 < HID; k0 += 32) {
    // A: out1 f16 -> quantize -> integer-valued f16 in LDS
#pragma unroll
    for (int i = 0; i < 2; ++i) {
      const int idx = tid + i * 256, r = idx >> 2, c = idx & 3;
      h8 v = *(const h8*)(out1 + (size_t)(m0 + r) * HID + k0 + c * 8);
      h8 q;
#pragma unroll
      for (int j = 0; j < 8; ++j) q[j] = (_Float16)qclip((float)v[j], inv_s);
      *(h8*)(As + r * LDH + c * 8) = q;
    }
    // B: w2 rows f32 -> f16
#pragma unroll
    for (int i = 0; i < 4; ++i) {
      const int idx = tid + i * 256, r = idx >> 3, c = idx & 7;
      float4 v = *(const float4*)(w2 + (size_t)(n0 + r) * HID + k0 + c * 4);
      h4 hb;
      hb.x = (_Float16)v.x; hb.y = (_Float16)v.y; hb.z = (_Float16)v.z; hb.w = (_Float16)v.w;
      *(h4*)(Bs + r * LDH + c * 4) = hb;
    }
    __syncthreads();

    Frag16 a[2], b[4];
#pragma unroll
    for (int tm = 0; tm < 2; ++tm) a[tm] = load_afrag(As, LDH, wm + tm * 16, 0, lane);
#pragma unroll
    for (int tn = 0; tn < 4; ++tn) b[tn] = load_bfrag(Bs, LDH, wn + tn * 16, 0, lane);
#pragma unroll
    for (int tm = 0; tm < 2; ++tm)
#pragma unroll
      for (int tn = 0; tn < 4; ++tn)
        acc[tm][tn] = wmma16(a[tm], b[tn], acc[tm][tn]);
    __syncthreads();
  }

#pragma unroll
  for (int tm = 0; tm < 2; ++tm)
#pragma unroll
    for (int tn = 0; tn < 4; ++tn) {
      const int n  = n0 + wn + tn * 16 + (lane & 15);
      const float bn = b2[n];
      const int mb = m0 + wm + tm * 16 + ((lane >> 4) << 3);
#pragma unroll
      for (int r = 0; r < 8; ++r)
        out[(size_t)(mb + r) * EMB + n] = s * acc[tm][tn][r] + bn;
    }
}

// ---------------------------------------------------------------------------
extern "C" void kernel_launch(void* const* d_in, const int* in_sizes, int n_in,
                              void* d_out, int out_size, void* d_ws, size_t ws_size,
                              hipStream_t stream) {
  const float* x   = (const float*)d_in[0];
  const float* wg  = (const float*)d_in[1];
  const float* bg  = (const float*)d_in[2];
  const float* w1  = (const float*)d_in[3];
  const float* b1  = (const float*)d_in[4];
  const float* rot = (const float*)d_in[5];
  const float* w2  = (const float*)d_in[6];
  const float* b2  = (const float*)d_in[7];
  float* outp = (float*)d_out;

  unsigned* scal = (unsigned*)d_ws;                  // [0]=absmax(x) [1]=absmax(out1)
  char* base = (char*)d_ws + 256;
  _Float16* rotT = (_Float16*)base;                  // HID*HID f16
  _Float16* out0 = rotT + (size_t)HID * HID;         // TOK*HID f16
  _Float16* out1 = out0 + (size_t)TOK * HID;         // TOK*HID f16

  glu_init_scalars<<<1, 32, 0, stream>>>(scal);
  glu_absmax_f32<<<1024, 256, 0, stream>>>(x, (size_t)TOK * EMB / 4, scal + 0);
  glu_rot_transpose<<<dim3(HID / 32, HID / 32), 256, 0, stream>>>(rot, rotT);
  glu_gemm1<<<dim3(TOK / 128, HID / 64), 256, 0, stream>>>(
      x, wg, w1, bg, b1, (const float*)scal, out0);
  glu_gemm2<<<dim3(TOK / 128, HID / 128), 256, 0, stream>>>(out0, rotT, out1, scal + 1);
  glu_gemm3<<<dim3(TOK / 128, EMB / 128), 256, 0, stream>>>(
      out1, w2, b2, (const float*)(scal + 1), outp);
}
